// MixtureOfBidders_44040594653617
// MI455X (gfx1250) — matrix operand order repaired and checked by reading, not compiled, not run on hardware
//
#include <hip/hip_runtime.h>

#define T_TOK   2048
#define H_DIM   2048
#define I_DIM   7168
#define E_NUM   8
#define R_RANK  64
#define SCAL    0.25f

typedef __attribute__((ext_vector_type(8)))  __bf16 v8bf;
typedef __attribute__((ext_vector_type(16))) __bf16 v16bf;
typedef __attribute__((ext_vector_type(8)))  float  v8f;
typedef int v4i __attribute__((vector_size(16)));  // b128 unit for async LDS copies

// ---------------------------------------------------------------------------
// gfx1250 async global->LDS copy (ASYNCcnt-tracked), guarded by __has_builtin.
// Builtin signature (from hipcc diagnostic): (AS1 v4i*, AS3 v4i*, imm, imm).
// ---------------------------------------------------------------------------
#if defined(__has_builtin)
#if __has_builtin(__builtin_amdgcn_global_load_async_to_lds_b128) && \
    __has_builtin(__builtin_amdgcn_s_wait_asynccnt)
#define USE_ASYNC_LDS 1
#endif
#endif

__device__ __forceinline__ void cp_g2l_b128(__bf16* lds_dst, const __bf16* g_src) {
#ifdef USE_ASYNC_LDS
  v4i* gs = const_cast<v4i*>(reinterpret_cast<const v4i*>(g_src));
  v4i* ls = reinterpret_cast<v4i*>(lds_dst);
  __builtin_amdgcn_global_load_async_to_lds_b128(
      (__attribute__((address_space(1))) v4i*)gs,
      (__attribute__((address_space(3))) v4i*)ls, 0, 0);
#else
  *reinterpret_cast<v8bf*>(lds_dst) = *reinterpret_cast<const v8bf*>(g_src);
#endif
}

__device__ __forceinline__ void async_wait_keep2() {
#ifdef USE_ASYNC_LDS
  __builtin_amdgcn_s_wait_asynccnt(2);
#endif
}
__device__ __forceinline__ void async_wait_all() {
#ifdef USE_ASYNC_LDS
  __builtin_amdgcn_s_wait_asynccnt(0);
#endif
}

// ---------------------------------------------------------------------------
// WMMA helpers (gfx1250: V_WMMA_F32_16X16X32_BF16, wave32)
// ---------------------------------------------------------------------------
__device__ __forceinline__ v8f wmma_bf16(v16bf a, v16bf b, v8f c) {
  return __builtin_amdgcn_wmma_f32_16x16x32_bf16(false, a, false, b, (short)0, c,
                                                 false, false);
}

// A (16x32 bf16, MxK): lanes 0-15 row M=lane, K chunks [k0..k0+7],[k0+16..k0+23];
// lanes 16-31 same rows, K chunks [k0+8..15],[k0+24..31].
__device__ __forceinline__ v16bf load_a_frag(const __bf16* rowPtr, int k0, int half) {
  v8bf c1 = *reinterpret_cast<const v8bf*>(rowPtr + k0 + half * 8);
  v8bf c2 = *reinterpret_cast<const v8bf*>(rowPtr + k0 + 16 + half * 8);
  v16bf a;
#pragma unroll
  for (int i = 0; i < 8; ++i) { a[i] = c1[i]; a[8 + i] = c2[i]; }
  return a;
}

// Same fragment sourced from an LDS-resident 32-wide row (ds_load_b128 x2).
__device__ __forceinline__ v16bf load_a_frag_lds(const __bf16* rowPtr, int half) {
  v8bf c1 = *reinterpret_cast<const v8bf*>(rowPtr + half * 8);
  v8bf c2 = *reinterpret_cast<const v8bf*>(rowPtr + 16 + half * 8);
  v16bf a;
#pragma unroll
  for (int i = 0; i < 8; ++i) { a[i] = c1[i]; a[8 + i] = c2[i]; }
  return a;
}

// B (32x16 bf16, KxN), weights row-major [N][K]: lane = column n, K chunks [k0+8h..+7],[k0+16+8h..+7].
__device__ __forceinline__ v16bf load_b_frag(const __bf16* colPtr, int k0, int half) {
  const __bf16* p = colPtr + k0 + half * 8;
  v8bf c1 = *reinterpret_cast<const v8bf*>(p);
  v8bf c2 = *reinterpret_cast<const v8bf*>(p + 16);
  v16bf b;
#pragma unroll
  for (int i = 0; i < 8; ++i) { b[i] = c1[i]; b[8 + i] = c2[i]; }
  return b;
}

// ---------------------------------------------------------------------------
// Big dense GEMM: C[M,N] f32 = A[M,K] * B[N,K]^T, M,N multiples of 128.
// Block = 8 waves; block tile 128(M) x 128(N); wave tile 64(M) x 32(N).
// A k-slab (128x32) double-buffered in LDS via async global->LDS copies.
// ---------------------------------------------------------------------------
#define AROW 40  // padded LDS row stride (elements) to spread banks

__global__ __launch_bounds__(256) void wmma_gemm_big(
    const __bf16* __restrict__ A, long lda,
    const __bf16* __restrict__ B, long ldb,
    float* __restrict__ C, long ldc, int K) {
  __shared__ __bf16 sA[2][128 * AROW];

  const int tid = threadIdx.x;
  const int wave = tid >> 5;
  const int lane = tid & 31;
  const int half = lane >> 4;
  const int r = lane & 15;
  const int mg = wave >> 2;                       // 0..1 (64-row m-group)
  const int ng = wave & 3;                        // 0..3 (32-col n-group)
  const int mBlock = blockIdx.y * 128;
  const int n0 = blockIdx.x * 128 + ng * 32;

  // per-thread staging slots: 2 x b128 per k-slab (512 chunks of 16B = 8KB)
  const int c0row = tid >> 2;                     // rows 0..63
  const int c0kc = (tid & 3) * 8;                 // k offset 0,8,16,24

  const __bf16* bp0 = B + (long)(n0 + r) * ldb;
  const __bf16* bp1 = B + (long)(n0 + 16 + r) * ldb;

  // prologue: stage k-slab 0 into buffer 0
  {
    const __bf16* a0 = A + (long)(mBlock + c0row) * lda + c0kc;
    cp_g2l_b128(&sA[0][c0row * AROW + c0kc], a0);
    const __bf16* a1 = A + (long)(mBlock + 64 + c0row) * lda + c0kc;
    cp_g2l_b128(&sA[0][(64 + c0row) * AROW + c0kc], a1);
  }

  v8f acc[4][2] = {};
  int p = 0;
  for (int k0 = 0; k0 < K; k0 += 32, p ^= 1) {
    const bool more = (k0 + 32) < K;
    if (more) {
      const int kn = k0 + 32;
      const __bf16* a0 = A + (long)(mBlock + c0row) * lda + kn + c0kc;
      cp_g2l_b128(&sA[p ^ 1][c0row * AROW + c0kc], a0);
      const __bf16* a1 = A + (long)(mBlock + 64 + c0row) * lda + kn + c0kc;
      cp_g2l_b128(&sA[p ^ 1][(64 + c0row) * AROW + c0kc], a1);
      async_wait_keep2();   // previous slab done; newest 2 still in flight
    } else {
      async_wait_all();
    }
    __syncthreads();

    v16bf b0 = load_b_frag(bp0, k0, half);
    v16bf b1 = load_b_frag(bp1, k0, half);
    const __bf16* sbase = &sA[p][0];
#pragma unroll
    for (int i = 0; i < 4; ++i) {
      const __bf16* ar = sbase + (mg * 64 + i * 16 + r) * AROW;
      v16bf a = load_a_frag_lds(ar, half);
      acc[i][0] = wmma_bf16(a, b0, acc[i][0]);
      acc[i][1] = wmma_bf16(a, b1, acc[i][1]);
    }
    __syncthreads();
  }

#pragma unroll
  for (int i = 0; i < 4; ++i) {
#pragma unroll
    for (int j = 0; j < 2; ++j) {
#pragma unroll
      for (int v = 0; v < 8; ++v) {
        int m = mBlock + mg * 64 + i * 16 + half * 8 + v;  // row = v + 8*(lane>>4)
        int col = n0 + j * 16 + r;
        C[(long)m * ldc + col] = acc[i][j][v];
      }
    }
  }
}

// ---------------------------------------------------------------------------
// Generic gathered WMMA GEMM. MODE 1: row-gather A / row-scatter C (bf16).
// MODE 3: row-gather A, scatter add scale*w*val into f32 C.
// ---------------------------------------------------------------------------
template <int MODE>
__global__ __launch_bounds__(256) void wmma_gemm(
    const __bf16* __restrict__ A, long lda, long aBatch, int aMask,
    const __bf16* __restrict__ B, long ldb, long bBatch, int bShift,
    float* __restrict__ Cf, __bf16* __restrict__ Cbf, long ldc, long cBatch, int cMask,
    int Mfix, int N, int K,
    const int* __restrict__ cnt, int cntStride,
    const int* __restrict__ lists, long listStride,
    const float* __restrict__ wlists, float scale) {
  const int z = blockIdx.z;
  const int Mloc = cnt ? cnt[z * cntStride] : Mfix;
  if (Mloc <= 0) return;
  const int wave = threadIdx.x >> 5;
  const int lane = threadIdx.x & 31;
  const int half = lane >> 4;
  const int r = lane & 15;
  const int m0 = (blockIdx.y * 2 + (wave >> 2)) * 32;
  const int n0 = (blockIdx.x * 4 + (wave & 3)) * 32;
  if (m0 >= Mloc || n0 >= N) return;

  const __bf16* Ab = A + (long)(z & aMask) * aBatch;
  const __bf16* Bb = B + (long)(z >> bShift) * bBatch;
  const int* lst = (MODE == 0) ? nullptr : (lists + (long)z * listStride);
  const float* wlt = (MODE == 3) ? (wlists + (long)z * listStride) : nullptr;

  const __bf16* ap[2];
#pragma unroll
  for (int i = 0; i < 2; ++i) {
    int m = m0 + i * 16 + r;
    int mc = (m < Mloc) ? m : (Mloc - 1);
    int ridx = (MODE == 0) ? mc : lst[mc];
    ap[i] = Ab + (long)ridx * lda;
  }
  const __bf16* bp[2] = { Bb + (long)(n0 + r) * ldb, Bb + (long)(n0 + 16 + r) * ldb };

  v8f acc[2][2] = {};
  for (int k0 = 0; k0 < K; k0 += 32) {
    v16bf a0 = load_a_frag(ap[0], k0, half);
    v16bf a1 = load_a_frag(ap[1], k0, half);
    v16bf b0 = load_b_frag(bp[0], k0, half);
    v16bf b1 = load_b_frag(bp[1], k0, half);
    acc[0][0] = wmma_bf16(a0, b0, acc[0][0]);
    acc[0][1] = wmma_bf16(a0, b1, acc[0][1]);
    acc[1][0] = wmma_bf16(a1, b0, acc[1][0]);
    acc[1][1] = wmma_bf16(a1, b1, acc[1][1]);
  }

  __bf16* Cb2 = (MODE == 1) ? (Cbf + (long)(z & cMask) * cBatch) : nullptr;
#pragma unroll
  for (int i = 0; i < 2; ++i) {
#pragma unroll
    for (int j = 0; j < 2; ++j) {
#pragma unroll
      for (int v = 0; v < 8; ++v) {
        int m = m0 + i * 16 + half * 8 + v;
        if (m >= Mloc) continue;
        int col = n0 + j * 16 + r;
        float val = acc[i][j][v];
        if (MODE == 0) {
          Cf[(long)m * ldc + col] = val;
        } else if (MODE == 1) {
          int t = lst[m];
          Cb2[(long)t * ldc + col] = (__bf16)val;
        } else {  // MODE == 3
          int t = lst[m];
          float wgt = wlt[m];
          float* p = Cf + (long)t * ldc + col;
          *p = *p + scale * wgt * val;
        }
      }
    }
  }
}

// ---------------------------------------------------------------------------
// Fused LoRA gate/up delta (K=64 WMMA) + base add + silu*up + weighted mix.
// Slot-0 launch writes hidmix, slot-1 accumulates (token-disjoint lists).
// ---------------------------------------------------------------------------
__global__ __launch_bounds__(256) void wmma_hid(
    const __bf16* __restrict__ Axg,                       // [T][128] bf16
    const __bf16* __restrict__ gBb,                       // [E][I][64] bf16
    const __bf16* __restrict__ uBb,                       // [E][I][64] bf16
    const float* __restrict__ gateBase,                   // [T][I] f32
    const float* __restrict__ upBase,                     // [T][I] f32
    __bf16* __restrict__ hidg,                            // [T][I] bf16 (slot pre-offset)
    __bf16* __restrict__ hidmix,                          // [T][I] bf16
    const int* __restrict__ cnt,                          // pre-offset slot; idx e*2
    const int* __restrict__ lists,                        // pre-offset slot; stride 2T
    const float* __restrict__ wlists, int slotAdd, float scale) {
  const int e = blockIdx.z;
  const int Mloc = cnt[e * 2];
  if (Mloc <= 0) return;
  const int wave = threadIdx.x >> 5;
  const int lane = threadIdx.x & 31;
  const int half = lane >> 4;
  const int r = lane & 15;
  const int m0 = (blockIdx.y * 2 + (wave >> 2)) * 32;
  const int n0 = (blockIdx.x * 4 + (wave & 3)) * 32;
  if (m0 >= Mloc || n0 >= I_DIM) return;
  const int* lst = lists + (long)e * (2 * T_TOK);
  const float* wlt = wlists + (long)e * (2 * T_TOK);

  const __bf16* ap[2];
#pragma unroll
  for (int i = 0; i < 2; ++i) {
    int m = m0 + i * 16 + r;
    int mc = (m < Mloc) ? m : (Mloc - 1);
    ap[i] = Axg + (long)lst[mc] * 128;
  }
  const __bf16* gbase = gBb + (long)e * I_DIM * 64;
  const __bf16* ubase = uBb + (long)e * I_DIM * 64;
  const __bf16* gp[2] = { gbase + (long)(n0 + r) * 64, gbase + (long)(n0 + 16 + r) * 64 };
  const __bf16* bup[2] = { ubase + (long)(n0 + r) * 64, ubase + (long)(n0 + 16 + r) * 64 };

  v8f accG[2][2] = {};
  v8f accU[2][2] = {};
#pragma unroll
  for (int k0 = 0; k0 < 64; k0 += 32) {
    v16bf ag0 = load_a_frag(ap[0], k0, half);
    v16bf ag1 = load_a_frag(ap[1], k0, half);
    v16bf au0 = load_a_frag(ap[0] + 64, k0, half);
    v16bf au1 = load_a_frag(ap[1] + 64, k0, half);
    v16bf bg0 = load_b_frag(gp[0], k0, half);
    v16bf bg1 = load_b_frag(gp[1], k0, half);
    v16bf bu0 = load_b_frag(bup[0], k0, half);
    v16bf bu1 = load_b_frag(bup[1], k0, half);
    accG[0][0] = wmma_bf16(ag0, bg0, accG[0][0]);
    accG[0][1] = wmma_bf16(ag0, bg1, accG[0][1]);
    accG[1][0] = wmma_bf16(ag1, bg0, accG[1][0]);
    accG[1][1] = wmma_bf16(ag1, bg1, accG[1][1]);
    accU[0][0] = wmma_bf16(au0, bu0, accU[0][0]);
    accU[0][1] = wmma_bf16(au0, bu1, accU[0][1]);
    accU[1][0] = wmma_bf16(au1, bu0, accU[1][0]);
    accU[1][1] = wmma_bf16(au1, bu1, accU[1][1]);
  }

#pragma unroll
  for (int i = 0; i < 2; ++i) {
#pragma unroll
    for (int j = 0; j < 2; ++j) {
#pragma unroll
      for (int v = 0; v < 8; ++v) {
        int m = m0 + i * 16 + half * 8 + v;
        if (m >= Mloc) continue;
        int t = lst[m];
        float wgt = wlt[m];
        long o = (long)t * I_DIM + (n0 + j * 16 + r);
        float g = gateBase[o] + scale * accG[i][j][v];
        float u = upBase[o] + scale * accU[i][j][v];
        float hid = g * u / (1.0f + __expf(-g));  // silu(g) * u
        hidg[o] = (__bf16)hid;
        if (slotAdd) {
          hidmix[o] = (__bf16)((float)hidmix[o] + wgt * hid);
        } else {
          hidmix[o] = (__bf16)(wgt * hid);
        }
      }
    }
  }
}

// ---------------------------------------------------------------------------
// Routing: confidence sigmoid, bids, top-2, softmax, expert/slot list build.
// One wave32 per token.
// ---------------------------------------------------------------------------
__global__ __launch_bounds__(256) void routing_kernel(
    const float* __restrict__ x, const float* __restrict__ cw,
    const float* __restrict__ cb, const float* __restrict__ wealth,
    int* __restrict__ cnt, int* __restrict__ lists, float* __restrict__ wlists) {
  const int wave = threadIdx.x >> 5;
  const int lane = threadIdx.x & 31;
  const int t = blockIdx.x * 8 + wave;
  if (t >= T_TOK) return;
  float acc[E_NUM];
#pragma unroll
  for (int e = 0; e < E_NUM; ++e) acc[e] = 0.0f;
  const float* xr = x + (long)t * H_DIM;
  for (int h = lane; h < H_DIM; h += 32) {
    float xv = xr[h];
#pragma unroll
    for (int e = 0; e < E_NUM; ++e) acc[e] += xv * cw[e * H_DIM + h];
  }
#pragma unroll
  for (int e = 0; e < E_NUM; ++e) {
#pragma unroll
    for (int off = 16; off > 0; off >>= 1) acc[e] += __shfl_xor(acc[e], off, 32);
  }
  if (lane == 0) {
    float bid[E_NUM];
#pragma unroll
    for (int e = 0; e < E_NUM; ++e) {
      float c = 1.0f / (1.0f + __expf(-(acc[e] + cb[e])));
      bid[e] = c * wealth[e];
    }
    int e0 = 0; float b0 = bid[0];
#pragma unroll
    for (int e = 1; e < E_NUM; ++e) if (bid[e] > b0) { b0 = bid[e]; e0 = e; }
    int e1 = (e0 == 0) ? 1 : 0; float b1 = -3.4e38f;
#pragma unroll
    for (int e = 0; e < E_NUM; ++e)
      if (e != e0 && bid[e] > b1) { b1 = bid[e]; e1 = e; }
    float w0 = 1.0f / (1.0f + __expf(b1 - b0));
    float w1 = 1.0f - w0;
    int p0 = atomicAdd(&cnt[e0 * 2 + 0], 1);
    lists[(long)(e0 * 2 + 0) * T_TOK + p0] = t;
    wlists[(long)(e0 * 2 + 0) * T_TOK + p0] = w0;
    int p1 = atomicAdd(&cnt[e1 * 2 + 1], 1);
    lists[(long)(e1 * 2 + 1) * T_TOK + p1] = t;
    wlists[(long)(e1 * 2 + 1) * T_TOK + p1] = w1;
  }
}

// ---------------------------------------------------------------------------
// f32 -> bf16 conversions
// ---------------------------------------------------------------------------
__global__ __launch_bounds__(256) void cvt_kernel(const float* __restrict__ src,
                                                  __bf16* __restrict__ dst, long n) {
  long i = (long)blockIdx.x * blockDim.x + threadIdx.x;
  long stride = (long)gridDim.x * blockDim.x;
  for (; i < n; i += stride) dst[i] = (__bf16)src[i];
}

// gA/uA [E][64][H] -> AB [E][128][H] (gA rows 0-63, uA rows 64-127)
__global__ __launch_bounds__(256) void cvt_concat_kernel(const float* __restrict__ src,
                                                         __bf16* __restrict__ dst, long n,
                                                         long perBatch, long dstBatchStride,
                                                         long dstOff) {
  long i = (long)blockIdx.x * blockDim.x + threadIdx.x;
  long stride = (long)gridDim.x * blockDim.x;
  for (; i < n; i += stride) {
    long b = i / perBatch;
    long rem = i - b * perBatch;
    dst[b * dstBatchStride + dstOff + rem] = (__bf16)src[i];
  }
}

// ---------------------------------------------------------------------------
extern "C" void kernel_launch(void* const* d_in, const int* in_sizes, int n_in,
                              void* d_out, int out_size, void* d_ws, size_t ws_size,
                              hipStream_t stream) {
  (void)in_sizes; (void)n_in; (void)out_size; (void)ws_size;
  const float* x  = (const float*)d_in[0];
  const float* cw = (const float*)d_in[1];
  const float* cb = (const float*)d_in[2];
  const float* we = (const float*)d_in[3];
  const float* wg = (const float*)d_in[4];
  const float* wu = (const float*)d_in[5];
  const float* wd = (const float*)d_in[6];
  const float* gA = (const float*)d_in[7];
  const float* gB = (const float*)d_in[8];
  const float* uA = (const float*)d_in[9];
  const float* uB = (const float*)d_in[10];
  const float* dA = (const float*)d_in[11];
  const float* dB = (const float*)d_in[12];
  float* out = (float*)d_out;

  char* base = (char*)d_ws;
  size_t off = 0;
  auto alloc = [&](size_t bytes) -> char* {
    char* p = base + off;
    off = (off + bytes + 255) & ~(size_t)255;
    return p;
  };
  __bf16* xb     = (__bf16*)alloc((size_t)T_TOK * H_DIM * 2);
  __bf16* wg_b   = (__bf16*)alloc((size_t)I_DIM * H_DIM * 2);
  __bf16* wu_b   = (__bf16*)alloc((size_t)I_DIM * H_DIM * 2);
  __bf16* wd_b   = (__bf16*)alloc((size_t)H_DIM * I_DIM * 2);
  __bf16* AB_b   = (__bf16*)alloc((size_t)E_NUM * 128 * H_DIM * 2);
  __bf16* gB_b   = (__bf16*)alloc((size_t)E_NUM * I_DIM * 64 * 2);
  __bf16* uB_b   = (__bf16*)alloc((size_t)E_NUM * I_DIM * 64 * 2);
  __bf16* dA_b   = (__bf16*)alloc((size_t)E_NUM * 64 * I_DIM * 2);
  __bf16* dB_b   = (__bf16*)alloc((size_t)E_NUM * H_DIM * 64 * 2);
  float*  gbase  = (float*)alloc((size_t)T_TOK * I_DIM * 4);
  float*  ubase  = (float*)alloc((size_t)T_TOK * I_DIM * 4);
  __bf16* hidmix = (__bf16*)alloc((size_t)T_TOK * I_DIM * 2);
  __bf16* xgau   = (__bf16*)alloc((size_t)2 * T_TOK * 128 * 2);
  __bf16* hidg   = (__bf16*)alloc((size_t)2 * T_TOK * I_DIM * 2);
  __bf16* hAb    = (__bf16*)alloc((size_t)2 * T_TOK * 64 * 2);
  int*    cnt    = (int*)alloc(256);
  int*    lists  = (int*)alloc((size_t)16 * T_TOK * 4);
  float*  wlists = (float*)alloc((size_t)16 * T_TOK * 4);

  const dim3 blk(256);
  const dim3 gcvt(4096);

  // 1) conversions f32 -> bf16
  cvt_kernel<<<gcvt, blk, 0, stream>>>(x, xb, (long)T_TOK * H_DIM);
  cvt_kernel<<<gcvt, blk, 0, stream>>>(wg, wg_b, (long)I_DIM * H_DIM);
  cvt_kernel<<<gcvt, blk, 0, stream>>>(wu, wu_b, (long)I_DIM * H_DIM);
  cvt_kernel<<<gcvt, blk, 0, stream>>>(wd, wd_b, (long)H_DIM * I_DIM);
  cvt_kernel<<<gcvt, blk, 0, stream>>>(gB, gB_b, (long)E_NUM * I_DIM * 64);
  cvt_kernel<<<gcvt, blk, 0, stream>>>(uB, uB_b, (long)E_NUM * I_DIM * 64);
  cvt_kernel<<<gcvt, blk, 0, stream>>>(dA, dA_b, (long)E_NUM * 64 * I_DIM);
  cvt_kernel<<<gcvt, blk, 0, stream>>>(dB, dB_b, (long)E_NUM * H_DIM * 64);
  cvt_concat_kernel<<<gcvt, blk, 0, stream>>>(gA, AB_b, (long)E_NUM * 64 * H_DIM,
                                              (long)64 * H_DIM, (long)128 * H_DIM, 0);
  cvt_concat_kernel<<<gcvt, blk, 0, stream>>>(uA, AB_b, (long)E_NUM * 64 * H_DIM,
                                              (long)64 * H_DIM, (long)128 * H_DIM,
                                              (long)64 * H_DIM);

  // 2) routing + expert/slot lists
  (void)hipMemsetAsync(cnt, 0, 64, stream);
  routing_kernel<<<dim3(T_TOK / 8), blk, 0, stream>>>(x, cw, cb, we, cnt, lists, wlists);

  // 3) per-(expert,slot) LoRA-A projections: xgau[s][t][0:128] = x_t @ [gA_e||uA_e]^T
  wmma_gemm<1><<<dim3(1, T_TOK / 64, 16), blk, 0, stream>>>(
      xb, H_DIM, 0, 0, AB_b, H_DIM, (long)128 * H_DIM, 1,
      nullptr, xgau, 128, (long)T_TOK * 128, 1,
      T_TOK, 128, H_DIM, cnt, 1, lists, T_TOK, nullptr, 0.0f);

  // 4) base gate/up GEMMs: [T,I] = xb @ W^T (async-LDS staged big GEMM)
  wmma_gemm_big<<<dim3(I_DIM / 128, T_TOK / 128, 1), blk, 0, stream>>>(
      xb, H_DIM, wg_b, H_DIM, gbase, I_DIM, H_DIM);
  wmma_gemm_big<<<dim3(I_DIM / 128, T_TOK / 128, 1), blk, 0, stream>>>(
      xb, H_DIM, wu_b, H_DIM, ubase, I_DIM, H_DIM);

  // 5) fused LoRA-B + silu*up + weighted mix; slot 0 writes, slot 1 accumulates
  for (int s = 0; s < 2; ++s) {
    wmma_hid<<<dim3(I_DIM / 128, T_TOK / 64, 8), blk, 0, stream>>>(
        xgau + (long)s * T_TOK * 128, gB_b, uB_b, gbase, ubase,
        hidg + (long)s * T_TOK * I_DIM, hidmix,
        cnt + s, lists + (long)s * T_TOK, wlists + (long)s * T_TOK, s, SCAL);
  }

  // 6) down-LoRA A: hA[s][t][0:64] = hid @ dA_e^T  (K = I)
  wmma_gemm<1><<<dim3(1, T_TOK / 64, 16), blk, 0, stream>>>(
      hidg, I_DIM, (long)T_TOK * I_DIM, 1, dA_b, I_DIM, (long)64 * I_DIM, 1,
      nullptr, hAb, 64, (long)T_TOK * 64, 1,
      T_TOK, 64, I_DIM, cnt, 1, lists, T_TOK, nullptr, 0.0f);

  // 7) base down GEMM on the weighted mix: out = hidmix @ down^T
  wmma_gemm_big<<<dim3(H_DIM / 128, T_TOK / 128, 1), blk, 0, stream>>>(
      hidmix, I_DIM, wd_b, I_DIM, out, H_DIM, I_DIM);

  // 8) down-LoRA B scatter-add: out[t] += SCAL * w * (hA @ dB_e^T); slot phases serialize
  for (int s = 0; s < 2; ++s) {
    wmma_gemm<3><<<dim3(H_DIM / 128, T_TOK / 64, 8), blk, 0, stream>>>(
        hAb + (long)s * T_TOK * 64, 64, 0, 0, dB_b, 64, (long)H_DIM * 64, 0,
        out, nullptr, H_DIM, 0, 0,
        T_TOK, H_DIM, 64,
        cnt + s, 2, lists + (long)s * T_TOK, (long)2 * T_TOK,
        wlists + (long)s * T_TOK, SCAL);
  }
}
